// TransformerEncoderGPT_10350871183423
// MI455X (gfx1250) — matrix operand order repaired and checked by reading, not compiled, not run on hardware
//
#include <hip/hip_runtime.h>
#include <hip/hip_bf16.h>

typedef __bf16 bf16;
typedef __attribute__((ext_vector_type(16))) __bf16 v16bf;
typedef __attribute__((ext_vector_type(8)))  __bf16 v8bf;
typedef __attribute__((ext_vector_type(8)))  float  v8f;

#define B_   2
#define L_   1024
#define D_   768
#define H_   12
#define HD_  64
#define NL_  2
#define DFF_ 3072
#define V_   50257
#define EOS_ 50256
#define TOK_ (B_ * L_)   // 2048

// ---------------------------------------------------------------------------
// WMMA helpers (gfx1250 16x16x32 bf16, f32 accumulate)
// ---------------------------------------------------------------------------
static __device__ __forceinline__ v8f wmma_bf16(v16bf a, v16bf b, v8f c) {
  return __builtin_amdgcn_wmma_f32_16x16x32_bf16(
      /*neg_a=*/false, a, /*neg_b=*/false, b,
      /*c_mod=*/(short)0, c, /*reuse_a=*/false, /*reuse_b=*/false);
}

// A-matrix 16x32 (MxK), row-major source, lda elements between rows.
// ISA layout: lane l holds row m=l&15; halves 0-7 -> K=(l<16?0:8)+j,
// halves 8-15 -> K=(l<16?16:24)+(j-8).  lda must be a multiple of 8.
static __device__ __forceinline__ v16bf load_a16x32(const bf16* p, int lda) {
  const int l  = threadIdx.x & 31;
  const int m  = l & 15;
  const int k0 = (l < 16) ? 0 : 8;
  const bf16* r0 = p + m * lda + k0;
  v8bf lo = *(const v8bf*)(r0);
  v8bf hi = *(const v8bf*)(r0 + 16);
  v16bf a;
#pragma unroll
  for (int j = 0; j < 8; ++j) { a[j] = lo[j]; a[8 + j] = hi[j]; }
  return a;
}

// B-matrix 32x16 (KxN) where this lane's 32 K-values are contiguous in memory.
// Caller passes lanep = base + n*ld (n = lane&15 column).  ISA layout: elems
// 0-7 -> K=(l<16?0:8)+j, elems 8-15 -> K=(l<16?16:24)+(j-8).
static __device__ __forceinline__ v16bf load_b_klinear(const bf16* lanep) {
  const int kb = ((threadIdx.x & 31) < 16) ? 0 : 8;
  v8bf lo = *(const v8bf*)(lanep + kb);
  v8bf hi = *(const v8bf*)(lanep + kb + 16);
  v16bf b;
#pragma unroll
  for (int j = 0; j < 8; ++j) { b[j] = lo[j]; b[8 + j] = hi[j]; }
  return b;
}

// B-matrix 32x16 with row-strided source: B(k,n) = lanep[k*ld] (lanep already
// offset by column n).
static __device__ __forceinline__ v16bf load_b_strided(const bf16* lanep, int ld) {
  const int kb = ((threadIdx.x & 31) < 16) ? 0 : 8;
  v16bf b;
#pragma unroll
  for (int j = 0; j < 16; ++j) b[j] = lanep[(long)(kb + j + (j & 8)) * ld];
  return b;
}

// ---------------------------------------------------------------------------
// Small prep kernels
// ---------------------------------------------------------------------------
__global__ void k_f32_to_bf16(const float* __restrict__ in, bf16* __restrict__ out, long n) {
  long i = (long)blockIdx.x * blockDim.x + threadIdx.x;
  long s = (long)gridDim.x * blockDim.x;
  for (; i < n; i += s) out[i] = (bf16)in[i];
}

// seg = inclusive cumsum(is_eos), rel = i - cummax(where(is_eos, i, 0))
__global__ void k_meta(const int* __restrict__ ids, int* __restrict__ seg,
                       int* __restrict__ rel) {
  int b = threadIdx.x;
  if (b >= B_) return;
  int s = 0, last = 0;
  for (int i = 0; i < L_; ++i) {
    int e = (ids[b * L_ + i] == EOS_) ? 1 : 0;
    if (e) last = i;
    s += e;
    seg[b * L_ + i] = s;
    rel[b * L_ + i] = i - last;
  }
}

__global__ void k_embed(const int* __restrict__ ids, const int* __restrict__ rel,
                        const float* __restrict__ tok, const float* __restrict__ pos,
                        float* __restrict__ x, bf16* __restrict__ xb) {
  const int row = blockIdx.x;  // 0..TOK_-1
  const int id  = ids[row];
  const int r   = rel[row];
  const float sc = 27.712812921102035f;  // sqrt(768)
  for (int d = threadIdx.x; d < D_; d += blockDim.x) {
    float v = tok[(long)id * D_ + d] * sc + pos[(long)r * D_ + d];
    x[(long)row * D_ + d]  = v;
    xb[(long)row * D_ + d] = (bf16)v;
  }
}

// ---------------------------------------------------------------------------
// Generic WMMA GEMM: C[M,N] = A[M,K] (bf16, lda) * W[N,K]^T (bf16, ldw) + bias
// ACT: 0 = none, 1 = exact GELU.  Block = 128 threads (4 waves); each wave
// owns a 32(M) x 64(N) strip: 2 A-tiles x 4 B-tiles -> 8 WMMAs per k-step.
// M must be a multiple of 32.
// ---------------------------------------------------------------------------
template <int ACT>
__global__ void k_gemm(const bf16* __restrict__ A, const bf16* __restrict__ W,
                       const float* __restrict__ bias,
                       float* __restrict__ Cf, bf16* __restrict__ Cb,
                       int M, int N, int K, int lda, int ldw, int ldc) {
  const int wid    = threadIdx.x >> 5;
  const int l      = threadIdx.x & 31;
  const int n_lane = l & 15;
  const int mbase  = (l < 16) ? 0 : 8;
  const int tile_m = blockIdx.y * 32;
  const int nwave  = blockIdx.x * 256 + wid * 64;
  if (nwave >= N) return;  // wave-uniform

  v8f acc0[4] = {}, acc1[4] = {};
  const bf16* Arow0 = A + (long)tile_m * lda;
  const bf16* Arow1 = Arow0 + (long)16 * lda;

  for (int k = 0; k < K; k += 32) {
    const v16bf a0 = load_a16x32(Arow0 + k, lda);
    const v16bf a1 = load_a16x32(Arow1 + k, lda);
#pragma unroll
    for (int t = 0; t < 4; ++t) {
      const int tn = nwave + t * 16;
      if (tn >= N) continue;  // wave-uniform
      int nc = tn + n_lane;
      if (nc >= N) nc = N - 1;  // clamp pad columns (not stored)
      const v16bf b = load_b_klinear(W + (long)nc * ldw + k);
      acc0[t] = wmma_bf16(a0, b, acc0[t]);
      acc1[t] = wmma_bf16(a1, b, acc1[t]);
    }
  }

#pragma unroll
  for (int t = 0; t < 4; ++t) {
    const int tn = nwave + t * 16;
    if (tn >= N) continue;
    const int nc = tn + n_lane;
    if (nc >= N) continue;  // per-lane column guard (stores only)
    const float bv = bias ? bias[nc] : 0.0f;
#pragma unroll
    for (int r = 0; r < 8; ++r) {
      float v0 = acc0[t][r] + bv;
      float v1 = acc1[t][r] + bv;
      if (ACT == 1) {
        v0 = 0.5f * v0 * (1.0f + erff(v0 * 0.70710678118654752f));
        v1 = 0.5f * v1 * (1.0f + erff(v1 * 0.70710678118654752f));
      }
      const long off0 = (long)(tile_m + mbase + r) * ldc + nc;
      const long off1 = (long)(tile_m + 16 + mbase + r) * ldc + nc;
      if (Cf) { Cf[off0] = v0; Cf[off1] = v1; }
      if (Cb) { Cb[off0] = (bf16)v0; Cb[off1] = (bf16)v1; }
    }
  }
}

// ---------------------------------------------------------------------------
// Flash attention with segment-causal mask. One wave per (b, h, 16-query
// tile); block = 4 waves.  qkv: bf16 [B, L, 3*D], out: bf16 [B, L, D].
//
// Softmax strategy: stage the masked/scaled f32 S-tile (16x32) through LDS to
// transpose C-layout -> A-layout.  After readback each lane owns one full
// query row (split with lane^16), so row max / row sum need exactly one
// shfl_xor(16) each and the running stats (mi, li) are per-lane scalars.
// alpha / 1/li are pushed back to the C-layout via a 16-float LDS array.
// ---------------------------------------------------------------------------
__global__ void k_attn(const bf16* __restrict__ qkv, const int* __restrict__ seg,
                       bf16* __restrict__ out) {
  const int wid   = threadIdx.x >> 5;
  const int l     = threadIdx.x & 31;
  const int n     = l & 15;   // C-layout column / A-layout row
  const int mbase = (l < 16) ? 0 : 8;
  const int qt = blockIdx.x * 4 + wid;  // 0..63
  const int h  = blockIdx.y;
  const int b  = blockIdx.z;
  const int q0 = qt * 16;
  const int ld = 3 * D_;  // 2304

  __shared__ __align__(16) float sbuf[4][16 * 32];
  __shared__ __align__(16) float abuf[4][16];
  float* sw = sbuf[wid];
  float* aw = abuf[wid];

  const bf16* qbase = qkv + (long)(b * L_ + q0) * ld + h * HD_;
  const v16bf aq0 = load_a16x32(qbase, ld);        // head dims 0..31
  const v16bf aq1 = load_a16x32(qbase + 32, ld);   // head dims 32..63

  int qseg_r[8];
#pragma unroll
  for (int r = 0; r < 8; ++r) qseg_r[r] = seg[b * L_ + q0 + mbase + r];

  // per-lane (row = n) online-softmax state
  float mi = -1e30f, li = 0.0f;
  v8f o[4] = {};

  const bf16* kall = qkv + (long)b * L_ * ld + D_ + h * HD_;
  const bf16* vall = qkv + (long)b * L_ * ld + 2 * D_ + h * HD_;

  for (int kk = 0; kk < L_; kk += 32) {
    // ---- S = Q * K^T for keys kk..kk+31 (two 16-key subtiles) ----
    const bf16* kb0 = kall + (long)(kk + n) * ld;
    const bf16* kb1 = kall + (long)(kk + 16 + n) * ld;
    v8f s0 = {}, s1 = {};
    s0 = wmma_bf16(aq0, load_b_klinear(kb0), s0);
    s0 = wmma_bf16(aq1, load_b_klinear(kb0 + 32), s0);
    s1 = wmma_bf16(aq0, load_b_klinear(kb1), s1);
    s1 = wmma_bf16(aq1, load_b_klinear(kb1 + 32), s1);

    // ---- scale + mask in C-layout, stage f32 tile in LDS ----
    const int kp0 = kk + n, kp1 = kk + 16 + n;
    const int ks0 = seg[b * L_ + kp0];
    const int ks1 = seg[b * L_ + kp1];
#pragma unroll
    for (int r = 0; r < 8; ++r) {
      const int qp = q0 + mbase + r;
      const float v0 = (kp0 <= qp && ks0 == qseg_r[r]) ? s0[r] * 0.125f : -1e30f;
      const float v1 = (kp1 <= qp && ks1 == qseg_r[r]) ? s1[r] * 0.125f : -1e30f;
      sw[(mbase + r) * 32 + n]      = v0;
      sw[(mbase + r) * 32 + 16 + n] = v1;
    }

    // ---- A-layout readback: this lane holds row n, 16 of its 32 cols ----
    const int k0 = (l < 16) ? 0 : 8;
    const float4 f0 = *(const float4*)(sw + n * 32 + k0);
    const float4 f1 = *(const float4*)(sw + n * 32 + k0 + 4);
    const float4 f2 = *(const float4*)(sw + n * 32 + k0 + 16);
    const float4 f3 = *(const float4*)(sw + n * 32 + k0 + 20);
    float sv[16] = {f0.x, f0.y, f0.z, f0.w, f1.x, f1.y, f1.z, f1.w,
                    f2.x, f2.y, f2.z, f2.w, f3.x, f3.y, f3.z, f3.w};

    float mx = sv[0];
#pragma unroll
    for (int j = 1; j < 16; ++j) mx = fmaxf(mx, sv[j]);
    mx = fmaxf(mx, __shfl_xor(mx, 16, 32));  // combine the two row halves

    const float nm = fmaxf(mi, mx);
    const float al = __expf(mi - nm);
    mi = nm;

    v16bf ap;
    float rs = 0.0f;
#pragma unroll
    for (int j = 0; j < 16; ++j) {
      const float e = __expf(sv[j] - nm);
      rs += e;
      ap[j] = (bf16)e;
    }
    rs += __shfl_xor(rs, 16, 32);
    li = li * al + rs;

    // ---- redistribute alpha (row-indexed) to C-layout via LDS ----
    aw[n] = al;  // lanes l and l^16 write identical values
    const float4 a_lo = *(const float4*)(aw + mbase);
    const float4 a_hi = *(const float4*)(aw + mbase + 4);
    const float alc[8] = {a_lo.x, a_lo.y, a_lo.z, a_lo.w,
                          a_hi.x, a_hi.y, a_hi.z, a_hi.w};
#pragma unroll
    for (int t = 0; t < 4; ++t)
#pragma unroll
      for (int r = 0; r < 8; ++r) o[t][r] *= alc[r];

    // ---- O += P * V (32 keys x 64 head dims) ----
    const bf16* vb = vall + (long)kk * ld + n;
#pragma unroll
    for (int t = 0; t < 4; ++t) {
      v16bf bv = load_b_strided(vb + t * 16, ld);
      o[t] = wmma_bf16(ap, bv, o[t]);
    }
  }

  // ---- final 1/li to C-layout, normalize, store ----
  aw[n] = 1.0f / li;
  const float4 i_lo = *(const float4*)(aw + mbase);
  const float4 i_hi = *(const float4*)(aw + mbase + 4);
  const float inv[8] = {i_lo.x, i_lo.y, i_lo.z, i_lo.w,
                        i_hi.x, i_hi.y, i_hi.z, i_hi.w};
#pragma unroll
  for (int r = 0; r < 8; ++r) {
    const long orow = (long)(b * L_ + q0 + mbase + r) * D_ + h * HD_;
#pragma unroll
    for (int t = 0; t < 4; ++t) out[orow + t * 16 + n] = (bf16)(o[t][r] * inv[r]);
  }
}

// ---------------------------------------------------------------------------
// x = LayerNorm(x + h) * w + b ; also emits bf16 copy.  One block per token.
// ---------------------------------------------------------------------------
__global__ void k_res_ln(const float* __restrict__ x, const float* __restrict__ h,
                         const float* __restrict__ w, const float* __restrict__ bia,
                         float* __restrict__ xo, bf16* __restrict__ xob) {
  const int row = blockIdx.x;
  const float* xr = x + (long)row * D_;
  const float* hr = h + (long)row * D_;
  float vals[3];
  float s = 0.f, ss = 0.f;
#pragma unroll
  for (int i = 0; i < 3; ++i) {
    const int d = threadIdx.x + i * 256;
    const float v = xr[d] + hr[d];
    vals[i] = v;
    s += v;
    ss += v * v;
  }
  __shared__ float rs[8], rss[8];
#pragma unroll
  for (int off = 16; off; off >>= 1) {
    s  += __shfl_xor(s, off, 32);
    ss += __shfl_xor(ss, off, 32);
  }
  if ((threadIdx.x & 31) == 0) {
    rs[threadIdx.x >> 5]  = s;
    rss[threadIdx.x >> 5] = ss;
  }
  __syncthreads();
  float ts = 0.f, tss = 0.f;
#pragma unroll
  for (int i = 0; i < 8; ++i) { ts += rs[i]; tss += rss[i]; }
  const float mean = ts * (1.0f / D_);
  const float var  = tss * (1.0f / D_) - mean * mean;
  const float rstd = rsqrtf(var + 1e-5f);
#pragma unroll
  for (int i = 0; i < 3; ++i) {
    const int d = threadIdx.x + i * 256;
    const float v = (vals[i] - mean) * rstd * w[d] + bia[d];
    xo[(long)row * D_ + d]  = v;
    xob[(long)row * D_ + d] = (bf16)v;
  }
}

// ---------------------------------------------------------------------------
// Host launcher
// ---------------------------------------------------------------------------
extern "C" void kernel_launch(void* const* d_in, const int* in_sizes, int n_in,
                              void* d_out, int out_size, void* d_ws, size_t ws_size,
                              hipStream_t stream) {
  (void)in_sizes; (void)n_in; (void)out_size; (void)ws_size;
  const int*   ids   = (const int*)d_in[0];
  const float* tok   = (const float*)d_in[1];
  const float* pos   = (const float*)d_in[2];
  const float* qkv_w = (const float*)d_in[3];
  const float* qkv_b = (const float*)d_in[4];
  const float* out_w = (const float*)d_in[5];
  const float* out_b = (const float*)d_in[6];
  const float* ln1w  = (const float*)d_in[7];
  const float* ln1b  = (const float*)d_in[8];
  const float* ln2w  = (const float*)d_in[9];
  const float* ln2b  = (const float*)d_in[10];
  const float* f1w   = (const float*)d_in[11];
  const float* f1b   = (const float*)d_in[12];
  const float* f2w   = (const float*)d_in[13];
  const float* f2b   = (const float*)d_in[14];
  float* logits = (float*)d_out;

  char* wp = (char*)d_ws;
  auto alloc = [&](size_t bytes) -> char* {
    char* p = wp;
    wp += (bytes + 255) & ~((size_t)255);
    return p;
  };
  bf16*  tok_bf  = (bf16*)alloc((size_t)V_ * D_ * 2);
  bf16*  qkvw_bf = (bf16*)alloc((size_t)NL_ * 3 * D_ * D_ * 2);
  bf16*  outw_bf = (bf16*)alloc((size_t)NL_ * D_ * D_ * 2);
  bf16*  f1w_bf  = (bf16*)alloc((size_t)NL_ * DFF_ * D_ * 2);
  bf16*  f2w_bf  = (bf16*)alloc((size_t)NL_ * D_ * DFF_ * 2);
  float* x       = (float*)alloc((size_t)TOK_ * D_ * 4);
  bf16*  xb      = (bf16*)alloc((size_t)TOK_ * D_ * 2);
  bf16*  qkvact  = (bf16*)alloc((size_t)TOK_ * 3 * D_ * 2);
  bf16*  attn    = (bf16*)alloc((size_t)TOK_ * D_ * 2);
  float* hbuf    = (float*)alloc((size_t)TOK_ * DFF_ * 4);
  bf16*  h1      = (bf16*)alloc((size_t)TOK_ * DFF_ * 2);
  int*   seg     = (int*)alloc((size_t)TOK_ * 4);
  int*   rel     = (int*)alloc((size_t)TOK_ * 4);

  // Weight down-converts (fp32 -> bf16), re-run every call (deterministic).
  k_f32_to_bf16<<<2048, 256, 0, stream>>>(tok,   tok_bf,  (long)V_ * D_);
  k_f32_to_bf16<<<1024, 256, 0, stream>>>(qkv_w, qkvw_bf, (long)NL_ * 3 * D_ * D_);
  k_f32_to_bf16<<<512,  256, 0, stream>>>(out_w, outw_bf, (long)NL_ * D_ * D_);
  k_f32_to_bf16<<<1024, 256, 0, stream>>>(f1w,   f1w_bf,  (long)NL_ * DFF_ * D_);
  k_f32_to_bf16<<<1024, 256, 0, stream>>>(f2w,   f2w_bf,  (long)NL_ * D_ * DFF_);

  k_meta<<<1, 32, 0, stream>>>(ids, seg, rel);
  k_embed<<<TOK_, 256, 0, stream>>>(ids, rel, tok, pos, x, xb);

  const dim3 gQKV((3 * D_ + 255) / 256, TOK_ / 32);
  const dim3 gD((D_ + 255) / 256, TOK_ / 32);
  const dim3 gFF((DFF_ + 255) / 256, TOK_ / 32);
  const dim3 gAttn(L_ / 64, H_, B_);

  for (int li = 0; li < NL_; ++li) {
    // qkv = x @ qw^T + qb   -> bf16 [TOK, 3D]
    k_gemm<0><<<gQKV, 128, 0, stream>>>(xb, qkvw_bf + (size_t)li * 3 * D_ * D_,
                                        qkv_b + (size_t)li * 3 * D_, nullptr, qkvact,
                                        TOK_, 3 * D_, D_, D_, D_, 3 * D_);
    // segment-causal flash attention
    k_attn<<<gAttn, 128, 0, stream>>>(qkvact, seg, attn);
    // o = attn @ ow^T + ob  -> fp32 [TOK, D]
    k_gemm<0><<<gD, 128, 0, stream>>>(attn, outw_bf + (size_t)li * D_ * D_,
                                      out_b + (size_t)li * D_, hbuf, nullptr,
                                      TOK_, D_, D_, D_, D_, D_);
    // x = LN(x + o)
    k_res_ln<<<TOK_, 256, 0, stream>>>(x, hbuf, ln1w + (size_t)li * D_,
                                       ln1b + (size_t)li * D_, x, xb);
    // h1 = gelu(x @ f1^T + b1) -> bf16 [TOK, DFF]
    k_gemm<1><<<gFF, 128, 0, stream>>>(xb, f1w_bf + (size_t)li * DFF_ * D_,
                                       f1b + (size_t)li * DFF_, nullptr, h1,
                                       TOK_, DFF_, D_, D_, D_, DFF_);
    // h2 = h1 @ f2^T + b2 -> fp32 [TOK, D]
    k_gemm<0><<<gD, 128, 0, stream>>>(h1, f2w_bf + (size_t)li * D_ * DFF_,
                                      f2b + (size_t)li * D_, hbuf, nullptr,
                                      TOK_, D_, DFF_, DFF_, DFF_, D_);
    // x = LN(x + h2)
    k_res_ln<<<TOK_, 256, 0, stream>>>(x, hbuf, ln2w + (size_t)li * D_,
                                       ln2b + (size_t)li * D_, x, xb);
  }

  // logits = x @ tok_emb^T  (fp32 out, N=50257 with column guards)
  const dim3 gV((V_ + 255) / 256, TOK_ / 32);
  k_gemm<0><<<gV, 128, 0, stream>>>(xb, tok_bf, nullptr, logits, nullptr,
                                    TOK_, V_, D_, D_, D_, V_);
}
